// CompleteEnzymeModel_14061722927716
// MI455X (gfx1250) — matrix operand order repaired and checked by reading, not compiled, not run on hardware
//
#include <hip/hip_runtime.h>
#include <hip/hip_bf16.h>
#include <stdint.h>

#define DD   128
#define BGR  256
#define NCLS 7
#define EPS_ 1e-5f

typedef __attribute__((ext_vector_type(16))) __bf16 v16bf;
typedef __attribute__((ext_vector_type(8)))  float  v8f;
typedef unsigned short u16;

union FragU { uint4 q[2]; v16bf v; };

__device__ __forceinline__ u16 f2bf(float f) {
    uint32_t u = __float_as_uint(f);
    uint32_t r = u + 0x7FFFu + ((u >> 16) & 1u);   // round-to-nearest-even
    return (u16)(r >> 16);
}

__device__ __forceinline__ float blockReduceSum(float v, float* red) {
    int t = threadIdx.x;
    red[t] = v; __syncthreads();
    for (int s = blockDim.x >> 1; s > 0; s >>= 1) {
        if (t < s) red[t] += red[t + s];
        __syncthreads();
    }
    float r = red[0]; __syncthreads();
    return r;
}

// ---------------- weight repack: f32 [K][NCOL] -> bf16 B-fragment layout ----------------
// frag idx = ((tn*KC + kc)*32 + lane)*8 + j ; VGPR j packs K = base+2j, base+2j+1
__global__ void wfrag_kernel(const float* __restrict__ W, uint32_t* __restrict__ F,
                             int K, int ncol) {
    int idx = blockIdx.x * blockDim.x + threadIdx.x;
    int total = (K * ncol) >> 1;
    if (idx >= total) return;
    int j    = idx & 7;
    int lane = (idx >> 3) & 31;
    int rest = idx >> 8;
    int KC   = K >> 5;
    int kc   = rest % KC;
    int tn   = rest / KC;
    int col  = tn * 16 + (lane & 15);
    int k0   = kc * 32 + (lane >> 4) * 16 + 2 * j;
    uint32_t lo = f2bf(W[(size_t)k0 * ncol + col]);
    uint32_t hi = f2bf(W[(size_t)(k0 + 1) * ncol + col]);
    F[idx] = lo | (hi << 16);
}

// ---------------- graph preprocessing ----------------
__global__ void deg_kernel(const int* __restrict__ src, const int* __restrict__ dst,
                           const int* __restrict__ batch, float* __restrict__ deg, int E) {
    int e = blockIdx.x * blockDim.x + threadIdx.x;
    if (e >= E) return;
    int s = src[e], d = dst[e];
    if (batch[s] == batch[d]) atomicAdd(&deg[d], 1.0f);
}

__global__ void dinv_kernel(const float* __restrict__ deg, float* __restrict__ dinv,
                            const int* __restrict__ batch, float* __restrict__ cnt, int N) {
    int n = blockIdx.x * blockDim.x + threadIdx.x;
    if (n >= N) return;
    dinv[n] = rsqrtf(1.0f + deg[n]);
    atomicAdd(&cnt[batch[n]], 1.0f);
}

__global__ void ew_kernel(const int* __restrict__ src, const int* __restrict__ dst,
                          const int* __restrict__ batch, const float* __restrict__ dinv,
                          float* __restrict__ ew, int E) {
    int e = blockIdx.x * blockDim.x + threadIdx.x;
    if (e >= E) return;
    int s = src[e], d = dst[e];
    ew[e] = (batch[s] == batch[d]) ? dinv[s] * dinv[d] : 0.0f;
}

// ---------------- embedding + LayerNorm -> bf16 ----------------
__global__ __launch_bounds__(DD) void embed_kernel(
    const float* __restrict__ x, const float* __restrict__ w, const float* __restrict__ b,
    const float* __restrict__ g, const float* __restrict__ be,
    u16* __restrict__ Hb, int N) {
    __shared__ float red[DD];
    int n = blockIdx.x, d = threadIdx.x;
    if (n >= N) return;
    float v = fmaxf(x[n] * w[d] + b[d], 0.0f);
    float m  = blockReduceSum(v, red) * (1.0f / DD);
    float dd = v - m;
    float var = blockReduceSum(dd * dd, red) * (1.0f / DD);
    Hb[(size_t)n * DD + d] = f2bf(dd * rsqrtf(var + EPS_) * g[d] + be[d]);
}

// ---------------- bf16 WMMA GEMM: C[N][ncol] = A[N][K] @ W, opt. bias+ReLU+stats ----------------
template<int K, bool EPI>
__global__ __launch_bounds__(256) void gemm_bf16(
    const u16* __restrict__ A, const uint32_t* __restrict__ Wfrag,
    float* __restrict__ C, const float* __restrict__ bias,
    const int* __restrict__ batch, float* __restrict__ sum, float* __restrict__ sumsq,
    int nrows, int ncol) {
    constexpr int KC = K / 32;
    __shared__ __align__(16) u16 As[32 * K];
    const int tid  = threadIdx.x;
    const int wave = tid >> 5, lane = tid & 31;
    const int rowbase = blockIdx.x * 32;
    if (rowbase >= nrows) return;

    // stage A tile: 32 consecutive rows are contiguous in memory
    {
        const uint4* s = (const uint4*)(A + (size_t)rowbase * K);
        uint4* p = (uint4*)As;
        #pragma unroll
        for (int i = tid; i < 32 * K / 8; i += 256) p[i] = s[i];
    }
    __syncthreads();

    const int tn  = blockIdx.y * 8 + wave;
    const int col = tn * 16 + (lane & 15);
    const int hi  = lane >> 4;
    const int r   = lane & 15;
    v8f c0 = {}; v8f c1 = {};
    const uint32_t* wp = Wfrag + ((size_t)tn * KC * 32 + lane) * 8;
    #pragma unroll
    for (int kc = 0; kc < KC; ++kc) {
        FragU b;
        b.q[0] = *(const uint4*)(wp);
        b.q[1] = *(const uint4*)(wp + 4);
        wp += 256;
        const int k0 = kc * 32 + hi * 16;       // A layout matches B frag: lanes<16 K0..7, lanes>=16 K16..23
        const int k1 = kc * 32 + hi * 16 + 8;   //                         lanes<16 K8..15, lanes>=16 K24..31
        FragU a0, a1;
        a0.q[0] = *(const uint4*)&As[r * K + k0];
        a0.q[1] = *(const uint4*)&As[r * K + k1];
        a1.q[0] = *(const uint4*)&As[(16 + r) * K + k0];
        a1.q[1] = *(const uint4*)&As[(16 + r) * K + k1];
        c0 = __builtin_amdgcn_wmma_f32_16x16x32_bf16(false, a0.v, false, b.v, (short)0, c0, false, false);
        c1 = __builtin_amdgcn_wmma_f32_16x16x32_bf16(false, a1.v, false, b.v, (short)0, c1, false, false);
    }

    float bc = EPI ? bias[col] : 0.0f;
    #pragma unroll
    for (int i = 0; i < 8; ++i) {
        int row0 = rowbase + hi * 8 + i;
        int row1 = row0 + 16;
        float v0 = c0[i], v1 = c1[i];
        if (EPI) {
            v0 = fmaxf(v0 + bc, 0.0f);
            v1 = fmaxf(v1 + bc, 0.0f);
            int g0 = batch[row0], g1 = batch[row1];
            atomicAdd(&sum[(size_t)g0 * ncol + col], v0);
            atomicAdd(&sumsq[(size_t)g0 * ncol + col], v0 * v0);
            atomicAdd(&sum[(size_t)g1 * ncol + col], v1);
            atomicAdd(&sumsq[(size_t)g1 * ncol + col], v1 * v1);
        }
        C[(size_t)row0 * ncol + col] = v0;
        C[(size_t)row1 * ncol + col] = v1;
    }
}

// ---------------- GCN pieces ----------------
__global__ void selfinit_kernel(const float* __restrict__ HW, const float* __restrict__ dinv,
                                const float* __restrict__ gb, float* __restrict__ AGG, int N) {
    int idx = blockIdx.x * blockDim.x + threadIdx.x;
    if (idx >= N * DD) return;
    int n = idx >> 7, d = idx & 127;
    float di = dinv[n];
    AGG[idx] = HW[idx] * di * di + gb[d];
}

__global__ __launch_bounds__(256) void edgeagg_kernel(
    const int* __restrict__ src, const int* __restrict__ dst, const float* __restrict__ ew,
    const float* __restrict__ HW, float* __restrict__ AGG, int E) {
    int e = blockIdx.x * 8 + (threadIdx.x >> 5);
    int lane = threadIdx.x & 31;
    if (e >= E) return;
    float w = ew[e];
    if (w == 0.0f) return;
    int s = src[e], d = dst[e];
    const float4 hv = *(const float4*)&HW[(size_t)s * DD + lane * 4];
    float* ap = &AGG[(size_t)d * DD + lane * 4];
    atomicAdd(ap + 0, hv.x * w);
    atomicAdd(ap + 1, hv.y * w);
    atomicAdd(ap + 2, hv.z * w);
    atomicAdd(ap + 3, hv.w * w);
}

__global__ void relu_bf16_kernel(const float* __restrict__ AGG, u16* __restrict__ Hb, int total) {
    int idx = blockIdx.x * blockDim.x + threadIdx.x;
    if (idx >= total) return;
    Hb[idx] = f2bf(fmaxf(AGG[idx], 0.0f));
}

// ---------------- instance norm -> bf16 (conv1) ----------------
__global__ void instnorm_kernel(const float* __restrict__ F, const float* __restrict__ sum,
                                const float* __restrict__ sumsq, const float* __restrict__ cnt,
                                const int* __restrict__ batch, u16* __restrict__ out,
                                int N, int nchan) {
    int idx = blockIdx.x * blockDim.x + threadIdx.x;
    if (idx >= N * nchan) return;
    int n = idx / nchan, c = idx - n * nchan;
    int g = batch[n];
    float ct = fmaxf(cnt[g], 1.0f);
    float m  = sum[(size_t)g * nchan + c] / ct;
    float var = sumsq[(size_t)g * nchan + c] / ct - m * m;
    out[idx] = f2bf((F[idx] - m) * rsqrtf(var + EPS_));
}

// ---------------- instance norm + global avg pool (conv2) ----------------
__global__ void pool_kernel(const float* __restrict__ F, const float* __restrict__ sum,
                            const float* __restrict__ sumsq, const float* __restrict__ cnt,
                            const int* __restrict__ batch, float* __restrict__ Z,
                            int N, int nchan) {
    int idx = blockIdx.x * blockDim.x + threadIdx.x;
    if (idx >= N * nchan) return;
    int n = idx / nchan, c = idx - n * nchan;
    int g = batch[n];
    float ct = fmaxf(cnt[g], 1.0f);
    float m  = sum[(size_t)g * nchan + c] / ct;
    float var = sumsq[(size_t)g * nchan + c] / ct - m * m;
    atomicAdd(&Z[(size_t)g * nchan + c], (F[idx] - m) * rsqrtf(var + EPS_));
}

// ---------------- classifier head (per-graph block) ----------------
__global__ __launch_bounds__(128) void cls_kernel(
    const float* __restrict__ Z, const float* __restrict__ cnt,
    const float* __restrict__ w1, const float* __restrict__ b1,
    const float* __restrict__ g1, const float* __restrict__ be1,
    const float* __restrict__ w2, const float* __restrict__ b2,
    const float* __restrict__ g2, const float* __restrict__ be2,
    const float* __restrict__ w3, const float* __restrict__ b3,
    float* __restrict__ out) {
    __shared__ float zin[512];
    __shared__ float s1[128];
    __shared__ float s2[64];
    __shared__ float red[128];
    int b = blockIdx.x, t = threadIdx.x;
    float ct = fmaxf(cnt[b], 1.0f);
    for (int i = t; i < 512; i += 128) zin[i] = Z[(size_t)b * 512 + i] / ct;
    __syncthreads();

    float acc = b1[t];
    for (int c = 0; c < 512; ++c) acc += zin[c] * w1[(size_t)c * 128 + t];
    acc = fmaxf(acc, 0.0f);
    float m = blockReduceSum(acc, red) * (1.0f / 128.0f);
    float d = acc - m;
    float v = blockReduceSum(d * d, red) * (1.0f / 128.0f);
    s1[t] = d * rsqrtf(v + EPS_) * g1[t] + be1[t];
    __syncthreads();

    float a2 = 0.0f;
    if (t < 64) {
        a2 = b2[t];
        for (int c = 0; c < 128; ++c) a2 += s1[c] * w2[(size_t)c * 64 + t];
        a2 = fmaxf(a2, 0.0f);
    }
    float m2 = blockReduceSum(t < 64 ? a2 : 0.0f, red) * (1.0f / 64.0f);
    float d2 = a2 - m2;
    float v2 = blockReduceSum(t < 64 ? d2 * d2 : 0.0f, red) * (1.0f / 64.0f);
    if (t < 64) s2[t] = d2 * rsqrtf(v2 + EPS_) * g2[t] + be2[t];
    __syncthreads();

    if (t < NCLS) {
        float o = b3[t];
        for (int c = 0; c < 64; ++c) o += s2[c] * w3[(size_t)c * NCLS + t];
        out[(size_t)b * NCLS + t] = o;
    }
}

// ================= launcher =================
extern "C" void kernel_launch(void* const* d_in, const int* in_sizes, int n_in,
                              void* d_out, int out_size, void* d_ws, size_t ws_size,
                              hipStream_t stream) {
    const float* x       = (const float*)d_in[0];
    const int*   batch   = (const int*)d_in[1];
    const int*   eidx    = (const int*)d_in[2];
    const float* w_emb   = (const float*)d_in[3];
    const float* b_emb   = (const float*)d_in[4];
    const float* g_emb   = (const float*)d_in[5];
    const float* be_emb  = (const float*)d_in[6];
    const float* gcn_w   = (const float*)d_in[7];
    const float* gcn_b   = (const float*)d_in[8];
    const float* conv1_w = (const float*)d_in[9];
    const float* conv1_b = (const float*)d_in[10];
    const float* conv2_w = (const float*)d_in[11];
    const float* conv2_b = (const float*)d_in[12];
    const float* cls_w1  = (const float*)d_in[13];
    const float* cls_b1  = (const float*)d_in[14];
    const float* cls_g1  = (const float*)d_in[15];
    const float* cls_be1 = (const float*)d_in[16];
    const float* cls_w2  = (const float*)d_in[17];
    const float* cls_b2  = (const float*)d_in[18];
    const float* cls_g2  = (const float*)d_in[19];
    const float* cls_be2 = (const float*)d_in[20];
    const float* cls_w3  = (const float*)d_in[21];
    const float* cls_b3  = (const float*)d_in[22];

    const int N = in_sizes[0];
    const int E = in_sizes[2] / 2;
    const int* src = eidx;
    const int* dst = eidx + E;

    // ---- workspace arena ----
    char* ws = (char*)d_ws;
    size_t off = 0;
    auto take = [&](size_t bytes) -> char* {
        char* p = ws + off;
        off += (bytes + 255) & ~(size_t)255;
        return p;
    };
    uint32_t* frag_gcn = (uint32_t*)take((size_t)4 * DD * DD / 2 * 4);
    uint32_t* frag_c1  = (uint32_t*)take((size_t)DD * 256 / 2 * 4);
    uint32_t* frag_c2  = (uint32_t*)take((size_t)256 * 512 / 2 * 4);
    float* ew   = (float*)take((size_t)E * 4);
    float* deg  = (float*)take((size_t)N * 4);
    float* dinv = (float*)take((size_t)N * 4);
    float* cnt  = (float*)take((size_t)BGR * 4);
    float* sum1 = (float*)take((size_t)BGR * 256 * 4);
    float* sq1  = (float*)take((size_t)BGR * 256 * 4);
    float* sum2 = (float*)take((size_t)BGR * 512 * 4);
    float* sq2  = (float*)take((size_t)BGR * 512 * 4);
    float* Z    = (float*)take((size_t)BGR * 512 * 4);
    u16*   Hb   = (u16*)take((size_t)N * DD * 2);
    float* HW   = (float*)take((size_t)N * DD * 4);
    float* AGG  = (float*)take((size_t)N * DD * 4);   // contiguous after HW
    float* F1   = HW;                                  // alias HW+AGG (dead after GCN stack)
    u16*   F1b  = (u16*)take((size_t)N * 256 * 2);
    float* F2   = (float*)take((size_t)N * 512 * 4);

    // ---- zero atomic targets (deterministic each call) ----
    hipMemsetAsync(deg,  0, (size_t)N * 4, stream);
    hipMemsetAsync(cnt,  0, (size_t)BGR * 4, stream);
    hipMemsetAsync(sum1, 0, (size_t)BGR * 256 * 4, stream);
    hipMemsetAsync(sq1,  0, (size_t)BGR * 256 * 4, stream);
    hipMemsetAsync(sum2, 0, (size_t)BGR * 512 * 4, stream);
    hipMemsetAsync(sq2,  0, (size_t)BGR * 512 * 4, stream);
    hipMemsetAsync(Z,    0, (size_t)BGR * 512 * 4, stream);

    // ---- weight repack to WMMA fragments ----
    for (int j = 0; j < 4; ++j)
        wfrag_kernel<<<(DD * DD / 2 + 255) / 256, 256, 0, stream>>>(
            gcn_w + (size_t)j * DD * DD, frag_gcn + (size_t)j * DD * DD / 2, DD, DD);
    wfrag_kernel<<<(DD * 256 / 2 + 255) / 256, 256, 0, stream>>>(conv1_w, frag_c1, DD, 256);
    wfrag_kernel<<<(256 * 512 / 2 + 255) / 256, 256, 0, stream>>>(conv2_w, frag_c2, 256, 512);

    // ---- graph norm coefficients ----
    deg_kernel<<<(E + 255) / 256, 256, 0, stream>>>(src, dst, batch, deg, E);
    dinv_kernel<<<(N + 255) / 256, 256, 0, stream>>>(deg, dinv, batch, cnt, N);
    ew_kernel<<<(E + 255) / 256, 256, 0, stream>>>(src, dst, batch, dinv, ew, E);

    // ---- node embedding ----
    embed_kernel<<<N, DD, 0, stream>>>(x, w_emb, b_emb, g_emb, be_emb, Hb, N);

    // ---- 4 GCN propagation layers ----
    const int rowblocks = (N + 31) / 32;
    for (int j = 0; j < 4; ++j) {
        gemm_bf16<DD, false><<<dim3(rowblocks, 1), 256, 0, stream>>>(
            Hb, frag_gcn + (size_t)j * DD * DD / 2, HW,
            nullptr, nullptr, nullptr, nullptr, N, DD);
        selfinit_kernel<<<(N * DD + 255) / 256, 256, 0, stream>>>(
            HW, dinv, gcn_b + (size_t)j * DD, AGG, N);
        edgeagg_kernel<<<(E + 7) / 8, 256, 0, stream>>>(src, dst, ew, HW, AGG, E);
        relu_bf16_kernel<<<(N * DD + 255) / 256, 256, 0, stream>>>(AGG, Hb, N * DD);
    }

    // ---- conv1 (bias+ReLU+stats fused) -> instance norm -> bf16 ----
    gemm_bf16<DD, true><<<dim3(rowblocks, 2), 256, 0, stream>>>(
        Hb, frag_c1, F1, conv1_b, batch, sum1, sq1, N, 256);
    instnorm_kernel<<<((size_t)N * 256 + 255) / 256, 256, 0, stream>>>(
        F1, sum1, sq1, cnt, batch, F1b, N, 256);

    // ---- conv2 -> instance norm + global avg pool ----
    gemm_bf16<256, true><<<dim3(rowblocks, 4), 256, 0, stream>>>(
        F1b, frag_c2, F2, conv2_b, batch, sum2, sq2, N, 512);
    pool_kernel<<<((size_t)N * 512 + 255) / 256, 256, 0, stream>>>(
        F2, sum2, sq2, cnt, batch, Z, N, 512);

    // ---- classifier head ----
    cls_kernel<<<BGR, 128, 0, stream>>>(Z, cnt,
        cls_w1, cls_b1, cls_g1, cls_be1,
        cls_w2, cls_b2, cls_g2, cls_be2,
        cls_w3, cls_b3, (float*)d_out);
}
